// DTME_39127152066963
// MI455X (gfx1250) — matrix operand order repaired and checked by reading, not compiled, not run on hardware
//
#include <hip/hip_runtime.h>
#include <utility>

#define N_SEQ 41
#define DIM   256
#define BATCH 4096
#define FF    2048
#define ROWS  (BATCH * N_SEQ)   // 167936
#define SA    264                // padded LDS row stride (elems) for 256-wide tiles
#define SH    2056               // padded LDS row stride (elems) for 2048-wide H buffer
#define BTILE (256 * SA)         // one W_s LDS buffer, in elems

typedef __attribute__((ext_vector_type(8)))  __bf16 v8bf;
typedef __attribute__((ext_vector_type(16))) __bf16 v16bf;
typedef __attribute__((ext_vector_type(8)))  float  v8f;

typedef __attribute__((address_space(3))) unsigned short lds_us;

__device__ inline unsigned short f2bf(float f) {
  unsigned int u = __float_as_uint(f);
  u += 0x7FFFu + ((u >> 16) & 1u);          // round-to-nearest-even
  return (unsigned short)(u >> 16);
}

__device__ inline v16bf ld16(const unsigned short* p0, const unsigned short* p1) {
  v8bf a = *(const v8bf*)p0;
  v8bf b = *(const v8bf*)p1;
  return __builtin_shufflevector(a, b, 0,1,2,3,4,5,6,7,8,9,10,11,12,13,14,15);
}

__device__ inline v8f wmma_bf16(v16bf a, v16bf b, v8f c) {
  return __builtin_amdgcn_wmma_f32_16x16x32_bf16(false, a, false, b, (short)0, c,
                                                 false, false);
}

__device__ inline v8f zero8() {
  v8f z;
#pragma unroll
  for (int e = 0; e < 8; ++e) z[e] = 0.f;
  return z;
}

// ---- CDNA5 async global->LDS copy (GLOBAL_LOAD_ASYNC_TO_LDS_B128, ASYNCcnt) ----
// IOFFSET applies to both the global and LDS address, so one base pair +
// immediate offsets copies a whole row with zero address arithmetic.

template <int BYTE_OFF>
__device__ inline void async_copy_b128_off(const unsigned short* g, unsigned short* l) {
  unsigned loff = (unsigned)(uintptr_t)(lds_us*)l;   // raw LDS byte offset
  asm volatile("global_load_async_to_lds_b128 %0, %1, off offset:%c2"
               :: "v"(loff), "v"(g), "i"(BYTE_OFF)
               : "memory");
}

template <int... Q>
__device__ inline void stage_row_impl(const unsigned short* g, unsigned short* l,
                                      std::integer_sequence<int, Q...>) {
  (async_copy_b128_off<Q * 16>(g, l), ...);
}

template <int NCHUNKS>
__device__ inline void stage_row(const unsigned short* g, unsigned short* l) {
  stage_row_impl(g, l, std::make_integer_sequence<int, NCHUNKS>{});
}

__device__ inline void wait_async0() {
#if __has_builtin(__builtin_amdgcn_s_wait_asynccnt)
  __builtin_amdgcn_s_wait_asynccnt(0);
#else
  asm volatile("s_wait_asynccnt 0" ::: "memory");
#endif
}

// ---------------- weight prep ----------------

__global__ void k_cvt_bf16(const float* __restrict__ in, unsigned short* __restrict__ out, int n) {
  int i = blockIdx.x * 256 + threadIdx.x;
  if (i < n) out[i] = f2bf(in[i]);
}

// WT[h][s][o][d] = w_h[d][o][s]  (bf16, B-operand-friendly layout)
__global__ void k_prep_convw(const float* __restrict__ w_in, const float* __restrict__ w_out,
                             unsigned short* __restrict__ wt) {
  int e = blockIdx.x;                 // (h, s, o)
  int d = threadIdx.x;
  int h = e / (N_SEQ * DIM);
  int r = e % (N_SEQ * DIM);
  int s = r / DIM;
  int o = r % DIM;
  const float* src = h ? w_out : w_in;
  float v = src[(size_t)d * (DIM * N_SEQ) + o * N_SEQ + s];
  wt[(size_t)e * DIM + d] = f2bf(v);
}

// ---------------- LayerNorm + bf16 pack (wave per row) ----------------

__global__ void k_ln_pack(const float* __restrict__ x, const float* __restrict__ g,
                          const float* __restrict__ be, unsigned short* __restrict__ out) {
  int wv = threadIdx.x >> 5, lane = threadIdx.x & 31;
  int row = blockIdx.x * 8 + wv;
  const float* xr = x + (size_t)row * DIM;
  float v[8]; float s = 0.f, sq = 0.f;
#pragma unroll
  for (int j = 0; j < 8; ++j) { v[j] = xr[lane + 32 * j]; s += v[j]; sq += v[j] * v[j]; }
#pragma unroll
  for (int m = 16; m >= 1; m >>= 1) { s += __shfl_xor(s, m, 32); sq += __shfl_xor(sq, m, 32); }
  float mu  = s * (1.f / 256.f);
  float var = sq * (1.f / 256.f) - mu * mu;
  float rs  = rsqrtf(var + 1e-5f);
  unsigned short* orow = out + (size_t)row * DIM;
#pragma unroll
  for (int j = 0; j < 8; ++j) {
    int c = lane + 32 * j;
    orow[c] = f2bf((v[j] - mu) * rs * g[c] + be[c]);
  }
}

// ---------------- circular-conv channel-mix as GEMM over batch ----------------
// grid = (32 M-tiles, 41 k).  out tile = 128 batches x 256 channels for one k.
// acc[b, o] += XH[b, i, :] @ WT[h][(k-i)%41][:, :]^T over i = 0..40.
// W_s tiles double-buffered in LDS with async prefetch; A-fragments stream
// straight from L2 (per-wave rows, reused in-register across 16 col tiles).

__global__ __launch_bounds__(256) void k_conv(const unsigned short* __restrict__ xh,
                                              const unsigned short* __restrict__ wt,
                                              const float* __restrict__ ent,
                                              float* __restrict__ x2) {
  extern __shared__ unsigned short smem[];   // 2 x (256 x SA) bf16 ping-pong
  int k  = blockIdx.y;
  int b0 = blockIdx.x * 128;
  int h  = (b0 >= BATCH / 2) ? 1 : 0;
  const unsigned short* W = wt + (size_t)h * N_SEQ * DIM * DIM;
  int tid = threadIdx.x;
  int wv = tid >> 5, lane = tid & 31, half = lane >> 4, mr = lane & 15;
  int m0 = wv * 16;

  v8f acc[16];
#pragma unroll
  for (int t = 0; t < 16; ++t) acc[t] = zero8();

  // per-lane global base for this wave's A rows (row = b0+m0+mr)
  const unsigned short* abase =
      xh + (size_t)(b0 + m0 + mr) * N_SEQ * DIM + half * 8;

  // prefetch W_{s(0)} into buffer 0   (s(0) = k)
  stage_row<32>(W + ((size_t)k * DIM + tid) * DIM, smem + tid * SA);
  wait_async0();
  __syncthreads();

  int p = 0;
  for (int i = 0; i < N_SEQ; ++i) {
    if (i + 1 < N_SEQ) {       // async prefetch next W_s into the other buffer
      int sn = k - (i + 1); if (sn < 0) sn += N_SEQ;
      stage_row<32>(W + ((size_t)sn * DIM + tid) * DIM,
                    smem + (p ^ 1) * BTILE + tid * SA);
    }
    const unsigned short* Bc = smem + p * BTILE;
    const unsigned short* ap = abase + (size_t)i * DIM;
#pragma unroll
    for (int ks = 0; ks < 8; ++ks) {
      int k0 = ks * 32;
      v16bf a = ld16(ap + k0, ap + k0 + 16);               // global (L2) A frag
#pragma unroll
      for (int ct = 0; ct < 16; ++ct) {
        v16bf b = ld16(Bc + (ct * 16 + mr) * SA + k0 + half * 8,
                       Bc + (ct * 16 + mr) * SA + k0 + half * 8 + 16);
        acc[ct] = wmma_bf16(a, b, acc[ct]);
      }
    }
    wait_async0();
    __syncthreads();
    p ^= 1;
  }
  // fused residual: x2[b,k,:] = ent[b,k,:] + sa
#pragma unroll
  for (int ct = 0; ct < 16; ++ct) {
#pragma unroll
    for (int r = 0; r < 8; ++r) {
      int row = m0 + r + half * 8;
      int b   = b0 + row;
      int col = ct * 16 + mr;
      size_t idx = ((size_t)b * N_SEQ + k) * DIM + col;
      x2[idx] = acc[ct][r] + ent[idx];
    }
  }
}

// ---------------- fused FFN: out = x2 + relu(x2h @ w1^T + b1) @ w2^T + b2 ----------------
// 64 rows per workgroup; H kept as bf16 in LDS (64 x 2048).

__global__ __launch_bounds__(256) void k_ffn(const unsigned short* __restrict__ x2h,
                                             const unsigned short* __restrict__ w1b,
                                             const unsigned short* __restrict__ w2b,
                                             const float* __restrict__ b1,
                                             const float* __restrict__ b2,
                                             const float* __restrict__ x2,
                                             float* __restrict__ out) {
  extern __shared__ unsigned short smem[];
  unsigned short* Ab = smem;             // 64 x SA bf16
  unsigned short* Hb = smem + 64 * SA;   // 64 x SH bf16
  int row0 = blockIdx.x * 64;
  int tid = threadIdx.x;
  int wv = tid >> 5, lane = tid & 31, half = lane >> 4, mr = lane & 15;
  int rt = wv & 3, chh = wv >> 2;
  int m0 = rt * 16;
  {   // async stage A: 64 rows x 512B
    int r = tid >> 2, seg = tid & 3;
    stage_row<8>(x2h + (size_t)(row0 + r) * DIM + seg * 64,
                 Ab + r * SA + seg * 64);
  }
  wait_async0();
  __syncthreads();

  // stage 1: H = relu(A @ W1^T + b1), processed in 256-wide chunks of FF
  for (int fc = 0; fc < 8; ++fc) {
    int f0 = fc * 256;
    v8f hc[8];
#pragma unroll
    for (int t = 0; t < 8; ++t) hc[t] = zero8();
#pragma unroll
    for (int ks = 0; ks < 8; ++ks) {
      int k0 = ks * 32;
      v16bf a = ld16(Ab + (m0 + mr) * SA + k0 + half * 8,
                     Ab + (m0 + mr) * SA + k0 + 16 + half * 8);
#pragma unroll
      for (int ct = 0; ct < 8; ++ct) {
        int f = f0 + chh * 128 + ct * 16 + mr;
        const unsigned short* wr = w1b + (size_t)f * DIM + k0 + half * 8;
        v16bf b = ld16(wr, wr + 16);
        hc[ct] = wmma_bf16(a, b, hc[ct]);
      }
    }
#pragma unroll
    for (int ct = 0; ct < 8; ++ct) {
      int fcol = f0 + chh * 128 + ct * 16 + mr;
      float bb = b1[fcol];
#pragma unroll
      for (int r = 0; r < 8; ++r) {
        int row = m0 + r + half * 8;
        float v = hc[ct][r] + bb;
        v = v > 0.f ? v : 0.f;
        Hb[(size_t)row * SH + fcol] = f2bf(v);
      }
    }
  }
  __syncthreads();

  // stage 2: out = H @ W2^T + b2 + x2
  v8f oc[8];
#pragma unroll
  for (int t = 0; t < 8; ++t) oc[t] = zero8();
  for (int ks = 0; ks < 64; ++ks) {
    int k0 = ks * 32;
    v16bf a = ld16(Hb + (m0 + mr) * SH + k0 + half * 8,
                   Hb + (m0 + mr) * SH + k0 + 16 + half * 8);
#pragma unroll
    for (int ct = 0; ct < 8; ++ct) {
      int o = chh * 128 + ct * 16 + mr;
      const unsigned short* wr = w2b + (size_t)o * FF + k0 + half * 8;
      v16bf b = ld16(wr, wr + 16);
      oc[ct] = wmma_bf16(a, b, oc[ct]);
    }
  }
#pragma unroll
  for (int ct = 0; ct < 8; ++ct) {
    int col = chh * 128 + ct * 16 + mr;
    float bb = b2[col];
#pragma unroll
    for (int r = 0; r < 8; ++r) {
      int row = row0 + m0 + r + half * 8;
      size_t idx = (size_t)row * DIM + col;
      out[idx] = oc[ct][r] + bb + x2[idx];
    }
  }
}

// ---------------- host launcher ----------------

extern "C" void kernel_launch(void* const* d_in, const int* in_sizes, int n_in,
                              void* d_out, int out_size, void* d_ws, size_t ws_size,
                              hipStream_t stream) {
  (void)in_sizes; (void)n_in; (void)out_size; (void)ws_size;
  const float* ent   = (const float*)d_in[0];
  const float* w_in  = (const float*)d_in[1];
  const float* w_out = (const float*)d_in[2];
  const float* w1    = (const float*)d_in[3];
  const float* b1    = (const float*)d_in[4];
  const float* w2    = (const float*)d_in[5];
  const float* b2    = (const float*)d_in[6];
  const float* g1    = (const float*)d_in[7];
  const float* be1   = (const float*)d_in[8];
  const float* g2    = (const float*)d_in[9];
  const float* be2   = (const float*)d_in[10];

  char* ws = (char*)d_ws;
  size_t off = 0;
  auto alloc = [&](size_t bytes) {
    void* p = ws + off;
    off += (bytes + 255) & ~(size_t)255;
    return p;
  };
  unsigned short* XH  = (unsigned short*)alloc((size_t)ROWS * DIM * 2);
  unsigned short* WT  = (unsigned short*)alloc((size_t)2 * N_SEQ * DIM * DIM * 2);
  float*          X2  = (float*)alloc((size_t)ROWS * DIM * 4);
  unsigned short* X2H = (unsigned short*)alloc((size_t)ROWS * DIM * 2);
  unsigned short* W1B = (unsigned short*)alloc((size_t)FF * DIM * 2);
  unsigned short* W2B = (unsigned short*)alloc((size_t)DIM * FF * 2);

  k_cvt_bf16<<<(FF * DIM + 255) / 256, 256, 0, stream>>>(w1, W1B, FF * DIM);
  k_cvt_bf16<<<(DIM * FF + 255) / 256, 256, 0, stream>>>(w2, W2B, DIM * FF);
  k_prep_convw<<<2 * N_SEQ * DIM, 256, 0, stream>>>(w_in, w_out, WT);
  k_ln_pack<<<ROWS / 8, 256, 0, stream>>>(ent, g1, be1, XH);

  size_t convSmem = (size_t)2 * BTILE * 2;               // ~264 KB of 320 KB WGP LDS
  k_conv<<<dim3(BATCH / 128, N_SEQ), 256, convSmem, stream>>>(XH, WT, ent, X2);

  k_ln_pack<<<ROWS / 8, 256, 0, stream>>>(X2, g2, be2, X2H);

  size_t ffnSmem = (size_t)(64 * SA + 64 * SH) * 2;      // ~297 KB
  k_ffn<<<ROWS / 64, 256, ffnSmem, stream>>>(X2H, W1B, W2B, b1, b2, X2, (float*)d_out);
}
